// One_Class_Classfication_19859928776967
// MI455X (gfx1250) — matrix operand order repaired and checked by reading, not compiled, not run on hardware
//
#include <hip/hip_runtime.h>
#include <math.h>

#define N_DRUG 20000
#define PROT   1512
#define NN     (N_DRUG + PROT)   // 21512
#define GCN_IN 1024
#define GAT_IN 512
#define DOUT   200

typedef _Float16 v16h __attribute__((ext_vector_type(16)));
typedef _Float16 v2h  __attribute__((ext_vector_type(2)));
typedef float    v8f  __attribute__((ext_vector_type(8)));

static inline int ceil_div(int a, int b) { return (a + b - 1) / b; }

// ============================================================================
// WMMA GEMM: C[M,N] = A[M,K] @ B[K,N] (+ optional bias[N]).
// f32 in/out, f16 multiply (v_wmma_f32_16x16x32_f16), f32 accumulate.
// Block = 256 threads = 8 waves; each wave owns 16 rows (128-row supertile)
// and NT 16-col tiles (A fragment reused NT times). B staged to LDS in
// fragment-ready f16 layout each 32-K step. Staging is branch-free:
//  - columns >= N are CLAMPED (their tiles are never stored, garbage is ok)
//  - rows >= K are zero-filled, but only on the single uniform tail step
// A loaded as 4 x float4 per step; 8-float groups clamped to K-8 when fully
// OOB (K is always a multiple of 8 -> safe, since staged B rows there are 0).
// NT <= 8 so accumulators + fragments fit without scratch spills.
// ============================================================================
template <int NT>
__global__ void wmma_gemm(const float* __restrict__ A, const float* __restrict__ B,
                          const float* __restrict__ bias, float* __restrict__ C,
                          int M, int N, int K) {
  __shared__ _Float16 sBf[NT * 32 * 16];     // [tn][lane][i] fragment-ready
  const int t    = threadIdx.x;
  const int wave = t >> 5;
  const int lane = t & 31;
  const bool hi  = lane >= 16;
  const int l16  = lane & 15;
  const int row0 = blockIdx.x * 128 + wave * 16;
  const int colBase = blockIdx.y * (NT * 16);

  int arow = row0 + l16; if (arow >= M) arow = M - 1;   // clamp; stores masked
  const float* Arow = A + (size_t)arow * K;

  v8f acc[NT];
  const v8f vzero = {};
#pragma unroll
  for (int i = 0; i < NT; ++i) acc[i] = vzero;

  const int steps = (K + 31) >> 5;
  for (int s = 0; s < steps; ++s) {
    const int k0 = s << 5;
    // ---- stage B[k0:k0+32, colBase:+NT*16] -> LDS f16 fragments.
    // Thread stages a k-pair per iteration -> one packed 32-bit LDS store.
    if (k0 + 32 <= K) {                       // uniform fast path: no K guard
      for (int idx = t; idx < 16 * NT * 16; idx += 256) {
        int klh = idx / (NT * 16);            // 0..15 : k-pair index
        int n   = idx - klh * (NT * 16);
        int kl  = klh << 1;
        int gn  = colBase + n; if (gn >= N) gn = N - 1;   // clamp, branch-free
        float v0 = B[(size_t)(k0 + kl) * N + gn];
        float v1 = B[(size_t)(k0 + kl + 1) * N + gn];
        int tn = n >> 4;
        int ln = (n & 15) + ((kl & 8) ? 16 : 0);
        *(v2h*)&sBf[(tn * 32 + ln) * 16 + (kl & 7) + ((kl & 16) ? 8 : 0)] = (v2h){(_Float16)v0, (_Float16)v1};
      }
    } else {                                  // single tail step: zero past K
      for (int idx = t; idx < 16 * NT * 16; idx += 256) {
        int klh = idx / (NT * 16);
        int n   = idx - klh * (NT * 16);
        int kl  = klh << 1;
        int gn  = colBase + n; if (gn >= N) gn = N - 1;
        float v0 = (k0 + kl     < K) ? B[(size_t)(k0 + kl) * N + gn]     : 0.f;
        float v1 = (k0 + kl + 1 < K) ? B[(size_t)(k0 + kl + 1) * N + gn] : 0.f;
        int tn = n >> 4;
        int ln = (n & 15) + ((kl & 8) ? 16 : 0);
        *(v2h*)&sBf[(tn * 32 + ln) * 16 + (kl & 7) + ((kl & 16) ? 8 : 0)] = (v2h){(_Float16)v0, (_Float16)v1};
      }
    }
    __syncthreads();

    // ---- A fragment: two 8-float groups, clamped if fully OOB (B rows are 0)
    int kb0 = k0 + (hi ? 8 : 0);      if (kb0 + 8 > K) kb0 = K - 8;
    int kb1 = k0 + 16 + (hi ? 8 : 0); if (kb1 + 8 > K) kb1 = K - 8;
    float4 a0 = *(const float4*)(Arow + kb0);
    float4 a1 = *(const float4*)(Arow + kb0 + 4);
    float4 a2 = *(const float4*)(Arow + kb1);
    float4 a3 = *(const float4*)(Arow + kb1 + 4);
    v16h af;
    af[0]  = (_Float16)a0.x; af[1]  = (_Float16)a0.y; af[2]  = (_Float16)a0.z; af[3]  = (_Float16)a0.w;
    af[4]  = (_Float16)a1.x; af[5]  = (_Float16)a1.y; af[6]  = (_Float16)a1.z; af[7]  = (_Float16)a1.w;
    af[8]  = (_Float16)a2.x; af[9]  = (_Float16)a2.y; af[10] = (_Float16)a2.z; af[11] = (_Float16)a2.w;
    af[12] = (_Float16)a3.x; af[13] = (_Float16)a3.y; af[14] = (_Float16)a3.z; af[15] = (_Float16)a3.w;

    // ---- NT matrix ops, A fragment reused
#pragma unroll
    for (int tn = 0; tn < NT; ++tn) {
      v16h bf = *(const v16h*)&sBf[(tn * 32 + lane) * 16];
      acc[tn] = __builtin_amdgcn_wmma_f32_16x16x32_f16(false, af, false, bf,
                                                       (short)0, acc[tn], false, false);
    }
    __syncthreads();
  }

  // ---- store (masked M/N tails), optional bias
#pragma unroll
  for (int tn = 0; tn < NT; ++tn) {
    int ccol = colBase + tn * 16 + l16;
    if (ccol < N) {
      float bv = bias ? bias[ccol] : 0.f;
#pragma unroll
      for (int r = 0; r < 8; ++r) {
        int crow = row0 + r + (hi ? 8 : 0);
        if (crow < M) C[(size_t)crow * N + ccol] = acc[tn][r] + bv;
      }
    }
  }
}

static void launch_gemm(const float* A, const float* B, const float* bias, float* C,
                        int M, int N, int K, hipStream_t s) {
  int gx = ceil_div(M, 128);
  if (N <= 224) {          // N = 200 path: 7 tiles per wave, 2 col-chunks
    dim3 grid(gx, ceil_div(N, 112));
    wmma_gemm<7><<<grid, 256, 0, s>>>(A, B, bias, C, M, N, K);
  } else {                 // N = 1024 path: 8 tiles per wave, grid.y chunks
    dim3 grid(gx, ceil_div(N, 128));
    wmma_gemm<8><<<grid, 256, 0, s>>>(A, B, bias, C, M, N, K);
  }
}

// ============================================================================
// GCN: symmetric-norm message passing with self loops (weight 1).
// ============================================================================
__global__ void k_deg_init(float* deg, int n) {
  int i = blockIdx.x * blockDim.x + threadIdx.x;
  if (i < n) deg[i] = 1.0f;               // self-loop weight
}
__global__ void k_edge_deg(const int* __restrict__ edge, const float* __restrict__ w,
                           float* deg, int E) {
  int e = blockIdx.x * blockDim.x + threadIdx.x;
  if (e >= E) return;
  int d = edge[2 * e + 1];
  __hip_atomic_fetch_add(&deg[d], w[e], __ATOMIC_RELAXED, __HIP_MEMORY_SCOPE_AGENT);
}
__global__ void k_dinv(float* deg, int n) {
  int i = blockIdx.x * blockDim.x + threadIdx.x;
  if (i >= n) return;
  float d = deg[i];
  deg[i] = (d > 0.f) ? rsqrtf(d) : 0.f;
}
__global__ void k_gcn_self(const float* __restrict__ xg, const float* __restrict__ dinv,
                           float* __restrict__ out, long total) {
  long t = (long)blockIdx.x * blockDim.x + threadIdx.x;
  if (t >= total) return;
  int i = (int)(t / DOUT);
  float s = dinv[i] * dinv[i];            // dinv[i]*1*dinv[i] self-loop norm
  out[t] = s * xg[t];
}
__global__ void k_gcn_scatter(const int* __restrict__ edge, const float* __restrict__ w,
                              const float* __restrict__ dinv, const float* __restrict__ xg,
                              float* __restrict__ out, int E) {
  int e = blockIdx.x;
  int s = edge[2 * e], d = edge[2 * e + 1];
  float norm = dinv[s] * w[e] * dinv[d];
  const float* xs = xg + (size_t)s * DOUT;
  float* od = out + (size_t)d * DOUT;
  for (int c = threadIdx.x; c < DOUT; c += blockDim.x)
    __hip_atomic_fetch_add(&od[c], norm * xs[c], __ATOMIC_RELAXED, __HIP_MEMORY_SCOPE_AGENT);
}

// ============================================================================
// GAT (heads=1): e = LeakyReLU_0.2(a_src.x'_src + a_dst.x'_dst), softmax over dst.
// ============================================================================
__global__ void k_gat_scores(const float* __restrict__ xa, const float* __restrict__ a_src,
                             const float* __restrict__ a_dst, float* ssrc, float* sdst,
                             float* mbuf, int n) {
  int i = blockIdx.x * blockDim.x + threadIdx.x;
  if (i >= n) return;
  float ss = 0.f, sd = 0.f;
  const float* x = xa + (size_t)i * DOUT;
  for (int c = 0; c < DOUT; ++c) { float v = x[c]; ss += v * a_src[c]; sd += v * a_dst[c]; }
  ssrc[i] = ss; sdst[i] = sd;
  float e = ss + sd; e = (e > 0.f) ? e : 0.2f * e;   // self-loop edge
  mbuf[i] = e;                                        // init running max
}
__global__ void k_gat_edge_max(const int* __restrict__ edge, const float* __restrict__ ssrc,
                               const float* __restrict__ sdst, float* mbuf, int E) {
  int e = blockIdx.x * blockDim.x + threadIdx.x;
  if (e >= E) return;
  int s = edge[2 * e], d = edge[2 * e + 1];
  float v = ssrc[s] + sdst[d]; v = (v > 0.f) ? v : 0.2f * v;
  __hip_atomic_fetch_max(&mbuf[d], v, __ATOMIC_RELAXED, __HIP_MEMORY_SCOPE_AGENT);
}
__global__ void k_gat_denom_init(const float* ssrc, const float* sdst,
                                 const float* mbuf, float* denom, int n) {
  int i = blockIdx.x * blockDim.x + threadIdx.x;
  if (i >= n) return;
  float e = ssrc[i] + sdst[i]; e = (e > 0.f) ? e : 0.2f * e;
  denom[i] = expf(e - mbuf[i]);
}
__global__ void k_gat_edge_denom(const int* __restrict__ edge, const float* __restrict__ ssrc,
                                 const float* __restrict__ sdst, const float* __restrict__ mbuf,
                                 float* denom, int E) {
  int e = blockIdx.x * blockDim.x + threadIdx.x;
  if (e >= E) return;
  int s = edge[2 * e], d = edge[2 * e + 1];
  float v = ssrc[s] + sdst[d]; v = (v > 0.f) ? v : 0.2f * v;
  __hip_atomic_fetch_add(&denom[d], expf(v - mbuf[d]), __ATOMIC_RELAXED, __HIP_MEMORY_SCOPE_AGENT);
}
__global__ void k_gat_self(const float* __restrict__ xa, const float* __restrict__ ssrc,
                           const float* __restrict__ sdst, const float* __restrict__ mbuf,
                           const float* __restrict__ denom, float* __restrict__ out, long total) {
  long t = (long)blockIdx.x * blockDim.x + threadIdx.x;
  if (t >= total) return;
  int i = (int)(t / DOUT);
  float e = ssrc[i] + sdst[i]; e = (e > 0.f) ? e : 0.2f * e;
  float alpha = expf(e - mbuf[i]) / denom[i];
  out[t] = alpha * xa[t];
}
__global__ void k_gat_scatter(const int* __restrict__ edge, const float* __restrict__ ssrc,
                              const float* __restrict__ sdst, const float* __restrict__ mbuf,
                              const float* __restrict__ denom, const float* __restrict__ xa,
                              float* __restrict__ out, int E) {
  int e = blockIdx.x;
  int s = edge[2 * e], d = edge[2 * e + 1];
  float v = ssrc[s] + sdst[d]; v = (v > 0.f) ? v : 0.2f * v;
  float alpha = expf(v - mbuf[d]) / denom[d];
  const float* xs = xa + (size_t)s * DOUT;
  float* od = out + (size_t)d * DOUT;
  for (int c = threadIdx.x; c < DOUT; c += blockDim.x)
    __hip_atomic_fetch_add(&od[c], alpha * xs[c], __ATOMIC_RELAXED, __HIP_MEMORY_SCOPE_AGENT);
}

// ============================================================================
// Column-wise (axis=0) log_softmax, in place. One block per column.
// (Per-column bias constants cancel here, so b_gcn/b_gat are skipped.)
// ============================================================================
__global__ void col_log_softmax(float* __restrict__ X, int nrows, int ncols) {
  __shared__ float red[256];
  int col = blockIdx.x, t = threadIdx.x;
  float m = -3.0e38f;
  for (int r = t; r < nrows; r += 256) m = fmaxf(m, X[(size_t)r * ncols + col]);
  red[t] = m; __syncthreads();
  for (int s = 128; s > 0; s >>= 1) { if (t < s) red[t] = fmaxf(red[t], red[t + s]); __syncthreads(); }
  float cmax = red[0]; __syncthreads();
  float sum = 0.f;
  for (int r = t; r < nrows; r += 256) sum += expf(X[(size_t)r * ncols + col] - cmax);
  red[t] = sum; __syncthreads();
  for (int s = 128; s > 0; s >>= 1) { if (t < s) red[t] += red[t + s]; __syncthreads(); }
  float lse = logf(red[0]) + cmax;
  for (int r = t; r < nrows; r += 256) X[(size_t)r * ncols + col] -= lse;
}

// ============================================================================
// Head: gather pair -> conv(16,1,3,5) pad(2,2) -> LeakyReLU(0.01) -> AvgPool2 ->
// tanh -> FC(6400,2). One block per sample.
// ============================================================================
__global__ void classifier_kernel(const float* __restrict__ gcn, const float* __restrict__ gat,
                                  const int* __restrict__ idx, const float* __restrict__ Wc,
                                  const float* __restrict__ bc, const float* __restrict__ Wout,
                                  const float* __restrict__ bout, float* __restrict__ out) {
  __shared__ float F[2][400];
  __shared__ float sWc[240];
  __shared__ float sbc[16];
  __shared__ float red0[256], red1[256];
  int b = blockIdx.x, t = threadIdx.x;
  int id = idx[b];
  int r = id / PROT, p = id % PROT;
  for (int c = t; c < 400; c += 256) {
    F[0][c] = (c < DOUT) ? gcn[(size_t)r * DOUT + c] : gat[(size_t)r * DOUT + (c - DOUT)];
    F[1][c] = (c < DOUT) ? gcn[(size_t)p * DOUT + c] : gat[(size_t)p * DOUT + (c - DOUT)];
  }
  for (int c = t; c < 240; c += 256) sWc[c] = Wc[c];
  if (t < 16) sbc[t] = bc[t];
  __syncthreads();
  float a0 = 0.f, a1 = 0.f;
  for (int j = t; j < 6400; j += 256) {     // flat = o*400 + hp*200 + wp (C,H,W order)
    int o = j / 400;
    int rem = j - o * 400;
    int hp = rem / 200;
    int wp = rem - hp * 200;
    float s = 0.f;
    for (int dh = 0; dh < 2; ++dh) {
      int h = 2 * hp + dh;
      for (int dw = 0; dw < 2; ++dw) {
        int w = 2 * wp + dw;
        float v = sbc[o];
#pragma unroll
        for (int kh = 0; kh < 3; ++kh) {
          int hin = h + kh - 2;
          if (hin < 0 || hin > 1) continue;
#pragma unroll
          for (int kw = 0; kw < 5; ++kw) {
            int win = w + kw - 2;
            if (win < 0 || win >= 400) continue;
            v += F[hin][win] * sWc[o * 15 + kh * 5 + kw];
          }
        }
        s += (v > 0.f) ? v : 0.01f * v;
      }
    }
    float tv = tanhf(s * 0.25f);            // avg of 4, then tanh
    a0 += tv * Wout[j * 2 + 0];
    a1 += tv * Wout[j * 2 + 1];
  }
  red0[t] = a0; red1[t] = a1; __syncthreads();
  for (int s = 128; s > 0; s >>= 1) {
    if (t < s) { red0[t] += red0[t + s]; red1[t] += red1[t + s]; }
    __syncthreads();
  }
  if (t == 0) {
    out[b * 2 + 0] = red0[0] + bout[0];
    out[b * 2 + 1] = red1[0] + bout[1];
  }
}

// ============================================================================
extern "C" void kernel_launch(void* const* d_in, const int* in_sizes, int n_in,
                              void* d_out, int out_size, void* d_ws, size_t ws_size,
                              hipStream_t stream) {
  const float* drug     = (const float*)d_in[0];
  const float* protein  = (const float*)d_in[1];
  const int*   gcn_edge = (const int*)d_in[2];
  const float* gcn_w    = (const float*)d_in[3];
  const float* gat_in   = (const float*)d_in[4];
  const int*   gat_edge = (const int*)d_in[5];
  const int*   idx      = (const int*)d_in[6];
  const float* W_pre    = (const float*)d_in[7];
  const float* b_pre    = (const float*)d_in[8];
  const float* W_gcn    = (const float*)d_in[9];
  /* d_in[10] b_gcn: cancels in axis-0 log_softmax */
  const float* W_gat    = (const float*)d_in[11];
  const float* a_src    = (const float*)d_in[12];
  const float* a_dst    = (const float*)d_in[13];
  /* d_in[14] b_gat: cancels */
  const float* W_conv   = (const float*)d_in[15];
  const float* b_conv   = (const float*)d_in[16];
  const float* W_out    = (const float*)d_in[17];
  const float* b_out    = (const float*)d_in[18];
  float* out = (float*)d_out;

  int E1 = in_sizes[3];
  int E2 = in_sizes[5] / 2;
  int Bn = in_sizes[6];
  (void)n_in; (void)out_size; (void)ws_size;

  float* ws = (float*)d_ws;
  size_t off = 0;
  auto alloc = [&](size_t n) { float* p = ws + off; off += (n + 63) & ~(size_t)63; return p; };
  float* prot  = alloc((size_t)PROT * GCN_IN);   // [1512,1024]
  float* xg    = alloc((size_t)NN * DOUT);       // GCN x' = gcn_in @ W_gcn
  float* xa    = alloc((size_t)NN * DOUT);       // GAT x' = GAT_input @ W_gat
  float* gcn   = alloc((size_t)NN * DOUT);       // GCN output -> log_softmax in place
  float* gat   = alloc((size_t)NN * DOUT);       // GAT output -> log_softmax in place
  float* dinv  = alloc(NN);
  float* ssrc  = alloc(NN);
  float* sdst  = alloc(NN);
  float* mbuf  = alloc(NN);
  float* denom = alloc(NN);

  // --- dense GEMMs (WMMA) ---
  launch_gemm(protein, W_pre, b_pre, prot, PROT, GCN_IN, 1512, stream);               // protein pre-proj
  launch_gemm(drug, W_gcn, nullptr, xg, N_DRUG, DOUT, GCN_IN, stream);                // drug rows
  launch_gemm(prot, W_gcn, nullptr, xg + (size_t)N_DRUG * DOUT, PROT, DOUT, GCN_IN, stream);
  launch_gemm(gat_in, W_gat, nullptr, xa, NN, DOUT, GAT_IN, stream);                  // GAT proj

  // --- GCN message passing ---
  k_deg_init<<<ceil_div(NN, 256), 256, 0, stream>>>(dinv, NN);
  k_edge_deg<<<ceil_div(E1, 256), 256, 0, stream>>>(gcn_edge, gcn_w, dinv, E1);
  k_dinv<<<ceil_div(NN, 256), 256, 0, stream>>>(dinv, NN);
  long tot = (long)NN * DOUT;
  k_gcn_self<<<(int)((tot + 255) / 256), 256, 0, stream>>>(xg, dinv, gcn, tot);
  k_gcn_scatter<<<E1, 128, 0, stream>>>(gcn_edge, gcn_w, dinv, xg, gcn, E1);

  // --- GAT message passing ---
  k_gat_scores<<<ceil_div(NN, 256), 256, 0, stream>>>(xa, a_src, a_dst, ssrc, sdst, mbuf, NN);
  k_gat_edge_max<<<ceil_div(E2, 256), 256, 0, stream>>>(gat_edge, ssrc, sdst, mbuf, E2);
  k_gat_denom_init<<<ceil_div(NN, 256), 256, 0, stream>>>(ssrc, sdst, mbuf, denom, NN);
  k_gat_edge_denom<<<ceil_div(E2, 256), 256, 0, stream>>>(gat_edge, ssrc, sdst, mbuf, denom, E2);
  k_gat_self<<<(int)((tot + 255) / 256), 256, 0, stream>>>(xa, ssrc, sdst, mbuf, denom, gat, tot);
  k_gat_scatter<<<E2, 128, 0, stream>>>(gat_edge, ssrc, sdst, mbuf, denom, xa, gat, E2);

  // --- axis-0 log_softmax, in place ---
  col_log_softmax<<<DOUT, 256, 0, stream>>>(gcn, NN, DOUT);
  col_log_softmax<<<DOUT, 256, 0, stream>>>(gat, NN, DOUT);

  // --- conv/pool/tanh/FC head ---
  classifier_kernel<<<Bn, 256, 0, stream>>>(gcn, gat, idx, W_conv, b_conv, W_out, b_out, out);
}
